// ContrastiveMemoryLearning_19207093747914
// MI455X (gfx1250) — matrix operand order repaired and checked by reading, not compiled, not run on hardware
//
#include <hip/hip_runtime.h>
#include <hip/hip_bf16.h>

typedef __bf16 bf16;
typedef bf16  v16bf __attribute__((ext_vector_type(16)));
typedef float v8f   __attribute__((ext_vector_type(8)));

#define ENC_OFF   0
#define PROJ_OFF  196608          // 3*256*256
#define PRED_OFF  294912          // + 3*256*128
#define LOSS_OFF  393216          // + 3*256*128
#define WPACK_DW  (3*1152*256)    // dwords: 3 modalities * (9*8*16 frags) * 256 dwords
#define WPACK_BYTES (WPACK_DW*4)

// Dynamic LDS layout for the conv kernel
#define XL_OFF    0               // bf16 x[pos=256][ci=256]   -> 131072 B
#define ZL_OFF    131072          // bf16 zeros[32]            -> 64 B
#define YL_OFF    131136          // bf16 y[co=256][p=256]     -> 131072 B
#define ST_OFF    262208          // float gsum[16],gsq[16],mu[16],rstd[16] -> 256 B
#define SMEM_BYTES 262464

static __device__ __forceinline__ unsigned short bfbits(float f) {
    bf16 h = (bf16)f;
    return __builtin_bit_cast(unsigned short, h);
}

struct ConvArgs {
    const float* x[3];
    const float* cbias[3];
    const float* gng[3];
    const float* gnb[3];
};

// ---------------------------------------------------------------------------
// Kernel 0: pack conv weights into WMMA A-fragment order (bf16 pairs).
// A fragment (16x32, M=co, K=ci-in-block at fixed (kh,kw)):
//   lane l: M = l%16 ; hi = l>=16
//   dword v   : K pair {2v, 2v+1} + (hi?16:0)
// Fragment id f = (kk*8 + cb)*16 + ct ; 256 dwords each, lane-major (lane*8+v)
// ---------------------------------------------------------------------------
__global__ void pack_weights_kernel(const float* __restrict__ w0,
                                    const float* __restrict__ w1,
                                    const float* __restrict__ w2,
                                    unsigned* __restrict__ wpack) {
    int idx = blockIdx.x * blockDim.x + threadIdx.x;
    if (idx >= WPACK_DW) return;
    int m  = idx / (1152 * 256);
    int r  = idx % (1152 * 256);
    int f  = r / 256;
    int lv = r % 256;
    int lane = lv >> 3, v = lv & 7;
    int ct = f & 15;
    int cb = (f >> 4) & 7;
    int kk = f >> 7;
    int hi = lane >> 4;
    int co = ct * 16 + (lane & 15);
    int K0 = 2 * v + (hi ? 16 : 0);
    int ci = cb * 32 + K0;
    int kh = kk / 3, kw = kk % 3;
    const float* w = (m == 0) ? w0 : (m == 1) ? w1 : w2;
    size_t base = ((size_t)co * 256 + ci) * 9 + kh * 3 + kw;   // OIHW
    float f0 = w[base];
    float f1 = w[base + 9];                                    // ci+1
    wpack[idx] = ((unsigned)bfbits(f1) << 16) | (unsigned)bfbits(f0);
}

// ---------------------------------------------------------------------------
// Kernel 1: fused Conv3x3 (implicit GEMM via v_wmma_f32_16x16x32_bf16)
//           + GroupNorm(16) + ReLU + global average pool.
// One block = one (modality, image). 256 threads = 8 waves.
// A (weights) and B (LDS image) fragments are double-buffered so LDS / L2
// latency overlaps WMMA execution instead of stalling on s_wait_dscnt 0.
// ---------------------------------------------------------------------------
__global__ __launch_bounds__(256) void conv_gn_pool_kernel(
        ConvArgs a, const unsigned* __restrict__ wpack, float* __restrict__ enc) {
    extern __shared__ char smem[];
    bf16*  xl   = (bf16*)(smem + XL_OFF);
    bf16*  zl   = (bf16*)(smem + ZL_OFF);
    bf16*  yl   = (bf16*)(smem + YL_OFF);
    float* gsum = (float*)(smem + ST_OFF);
    float* gsq  = gsum + 16;
    float* gmu  = gsum + 32;
    float* grs  = gsum + 48;

    const int bid  = blockIdx.x;
    const int m    = bid >> 8;
    const int b    = bid & 255;
    const int tid  = threadIdx.x;
    const int lane = tid & 31;
    const int wav  = tid >> 5;
    const int n    = lane & 15;     // N / pixel-column index within tile
    const int hi   = lane >> 4;     // K-half select

    if (tid < 32) gsum[tid] = 0.f;          // gsum+gsq
    if (tid < 32) zl[tid] = (bf16)0.f;      // zero B operand for padding

    // Stage input image into LDS as bf16 [pos][ci]
    {
        const float* xg = a.x[m] + (size_t)b * 65536;
        unsigned* xrow = (unsigned*)(xl + tid * 256);
        for (int ci = 0; ci < 256; ci += 2) {
            float f0 = xg[ci * 256 + tid];
            float f1 = xg[(ci + 1) * 256 + tid];
            xrow[ci >> 1] = ((unsigned)bfbits(f1) << 16) | (unsigned)bfbits(f0);
        }
    }
    __syncthreads();

    const unsigned* wp = wpack + (size_t)m * (1152 * 256) + lane * 8;
    const v16bf* zp = (const v16bf*)zl;
    const float* cbias = a.cbias[m];

    for (int tci = 0; tci < 2; ++tci) {
        const int ct = wav * 2 + tci;        // co-tile == GroupNorm group
        float s = 0.f, q = 0.f;
        for (int ph = 0; ph < 2; ++ph) {
            v8f acc[8];
#pragma unroll
            for (int i = 0; i < 8; ++i) acc[i] = (v8f){0,0,0,0,0,0,0,0};

            v16bf afr[2], bfr[2];
            // preload A fragment for kcb = 0 (kk=0, cb=0)
            afr[0] = *(const v16bf*)(wp + (ct << 8));

#pragma unroll 2
            for (int kcb = 0; kcb < 72; ++kcb) {
                const int kk = kcb >> 3, cb = kcb & 7;
                const int dh = kk / 3 - 1, dw = kk % 3 - 1;
                const int in_w = n + dw;
                const bool wok = (unsigned)in_w < 16u;
                const bf16* bbase = xl + cb * 32 + hi * 16;

                // prefetch next A fragment one (kk,cb) step ahead
                const int kcbn = (kcb < 71) ? kcb + 1 : kcb;
                afr[(kcb + 1) & 1] =
                    *(const v16bf*)(wp + (((kcbn * 16 + ct)) << 8));

                // preload first B fragment of this h-run
                {
                    const int in_h0 = ph * 8 + dh;
                    const bool ok0 = wok && ((unsigned)in_h0 < 16u);
                    const v16bf* bp0 = ok0
                        ? (const v16bf*)(bbase + (in_h0 * 16 + in_w) * 256)
                        : zp;
                    bfr[0] = *bp0;
                }
#pragma unroll
                for (int hh = 0; hh < 8; ++hh) {
                    if (hh < 7) {   // prefetch B for hh+1 before the WMMA on hh
                        const int in_h = ph * 8 + hh + 1 + dh;
                        const bool ok = wok && ((unsigned)in_h < 16u);
                        const v16bf* bp = ok
                            ? (const v16bf*)(bbase + (in_h * 16 + in_w) * 256)
                            : zp;
                        bfr[(hh + 1) & 1] = *bp;
                    }
                    acc[hh] = __builtin_amdgcn_wmma_f32_16x16x32_bf16(
                        false, afr[kcb & 1], false, bfr[hh & 1],
                        (short)0, acc[hh], false, false);
                }
            }
            // bias + stats + spill y (bf16) to LDS
#pragma unroll
            for (int r = 0; r < 8; ++r) {
                const int co = ct * 16 + hi * 8 + r;
                const float bias = cbias[co];
#pragma unroll
                for (int hh = 0; hh < 8; ++hh) {
                    const int h = ph * 8 + hh;
                    float v = acc[hh][r] + bias;
                    s += v; q += v * v;
                    yl[co * 256 + h * 16 + n] = (bf16)v;
                }
            }
        }
        // wave owns whole group: cross-lane reduce, plain store
        for (int off = 16; off > 0; off >>= 1) {
            s += __shfl_xor(s, off, 32);
            q += __shfl_xor(q, off, 32);
        }
        if (lane == 0) { gsum[ct] = s; gsq[ct] = q; }
    }
    __syncthreads();

    if (tid < 16) {
        float mu  = gsum[tid] * (1.f / 4096.f);
        float var = gsq[tid] * (1.f / 4096.f) - mu * mu;
        gmu[tid] = mu;
        grs[tid] = rsqrtf(var + 1e-5f);
    }
    __syncthreads();

    // normalize + affine + relu + spatial mean (thread c = channel)
    {
        const int c = tid;
        const float mu = gmu[c >> 4], rs = grs[c >> 4];
        const float gam = a.gng[m][c], bet = a.gnb[m][c];
        const bf16* row = yl + c * 256;
        float acc = 0.f;
        for (int i = 0; i < 256; ++i) {
            int p = (i + c) & 255;   // rotate to spread LDS banks
            float v = ((float)row[p] - mu) * rs * gam + bet;
            acc += fmaxf(v, 0.f);
        }
        enc[(size_t)m * 65536 + b * 256 + c] = acc * (1.f / 256.f);
    }
}

// ---------------------------------------------------------------------------
// Kernel 2: projector  h=relu(e@W1+b1); z=h@W2+b2; LayerNorm(z)
// ---------------------------------------------------------------------------
__global__ __launch_bounds__(256) void projector_kernel(
        const float* __restrict__ enc,
        const float* __restrict__ w1, const float* __restrict__ b1,
        const float* __restrict__ w2, const float* __restrict__ b2,
        const float* __restrict__ lng, const float* __restrict__ lnb,
        float* __restrict__ proj) {
    __shared__ float e[256];
    __shared__ float h[256];
    __shared__ float zsh[128];
    __shared__ float mu_s, rs_s;
    const int row = blockIdx.x, t = threadIdx.x;
    e[t] = enc[(size_t)row * 256 + t];
    __syncthreads();
    float acc = b1[t];
    for (int k = 0; k < 256; ++k) acc += e[k] * w1[k * 256 + t];
    h[t] = fmaxf(acc, 0.f);
    __syncthreads();
    float z = 0.f;
    if (t < 128) {
        z = b2[t];
        for (int k = 0; k < 256; ++k) z += h[k] * w2[k * 128 + t];
        zsh[t] = z;
    }
    __syncthreads();
    if (t == 0) {
        float s = 0.f, q = 0.f;
        for (int i = 0; i < 128; ++i) { float v = zsh[i]; s += v; q += v * v; }
        float mu = s * (1.f / 128.f);
        mu_s = mu;
        rs_s = rsqrtf(q * (1.f / 128.f) - mu * mu + 1e-5f);
    }
    __syncthreads();
    if (t < 128)
        proj[(size_t)row * 128 + t] = (z - mu_s) * rs_s * lng[t] + lnb[t];
}

// ---------------------------------------------------------------------------
// Kernel 3: predictor  p = relu(z@W1+b1)@W2+b2
// ---------------------------------------------------------------------------
__global__ __launch_bounds__(128) void predictor_kernel(
        const float* __restrict__ proj,
        const float* __restrict__ w1, const float* __restrict__ b1,
        const float* __restrict__ w2, const float* __restrict__ b2,
        float* __restrict__ pred) {
    __shared__ float z[128];
    __shared__ float h[128];
    const int row = blockIdx.x, t = threadIdx.x;
    z[t] = proj[(size_t)row * 128 + t];
    __syncthreads();
    float a = b1[t];
    for (int k = 0; k < 128; ++k) a += z[k] * w1[k * 128 + t];
    h[t] = fmaxf(a, 0.f);
    __syncthreads();
    float o = b2[t];
    for (int k = 0; k < 128; ++k) o += h[k] * w2[k * 128 + t];
    pred[(size_t)row * 128 + t] = o;
}

// ---------------------------------------------------------------------------
// Kernel 4: per-row inverse L2 norms of projected (768 rows)
// ---------------------------------------------------------------------------
__global__ void norms_kernel(const float* __restrict__ proj,
                             float* __restrict__ inv_norm) {
    int i = blockIdx.x * blockDim.x + threadIdx.x;
    if (i >= 768) return;
    const float* zr = proj + (size_t)i * 128;
    float s = 0.f;
    for (int k = 0; k < 128; ++k) s += zr[k] * zr[k];
    inv_norm[i] = 1.f / fmaxf(sqrtf(s), 1e-8f);
}

// ---------------------------------------------------------------------------
// Kernel 5: per-(pair,row) loss term: logsumexp(S[k,:]) - S[k,k]
//   (identical to the [pos|negs] concat in the reference)
// ---------------------------------------------------------------------------
__global__ __launch_bounds__(256) void loss_row_kernel(
        const float* __restrict__ proj, const float* __restrict__ inv_norm,
        float* __restrict__ rowloss) {
    const int blk = blockIdx.x;
    const int pair = blk >> 8;
    const int k = blk & 255;
    const int pi = (pair == 2) ? 1 : 0;
    const int pj = (pair == 0) ? 1 : 2;
    const int t = threadIdx.x;
    __shared__ float zi[128];
    __shared__ float srow[256];
    if (t < 128) zi[t] = proj[((size_t)pi * 256 + k) * 128 + t];
    __syncthreads();
    const float* zj = proj + ((size_t)pj * 256 + t) * 128;
    float d = 0.f;
    for (int c = 0; c < 128; ++c) d += zi[c] * zj[c];
    srow[t] = d * inv_norm[pi * 256 + k] * inv_norm[pj * 256 + t] * 10.f; // / TEMP
    __syncthreads();
    if (t == 0) {
        float mx = -3.402823e38f;
        for (int l = 0; l < 256; ++l) mx = fmaxf(mx, srow[l]);
        float se = 0.f;
        for (int l = 0; l < 256; ++l) se += expf(srow[l] - mx);
        rowloss[blk] = (mx + logf(se)) - srow[k];
    }
}

__global__ void loss_reduce_kernel(const float* __restrict__ rowloss,
                                   float* __restrict__ loss) {
    if (threadIdx.x == 0 && blockIdx.x == 0) {
        float s = 0.f;
        for (int i = 0; i < 768; ++i) s += rowloss[i];
        *loss = s * (1.f / (256.f * 3.f));
    }
}

// ---------------------------------------------------------------------------
extern "C" void kernel_launch(void* const* d_in, const int* in_sizes, int n_in,
                              void* d_out, int out_size, void* d_ws, size_t ws_size,
                              hipStream_t stream) {
    (void)in_sizes; (void)n_in; (void)out_size; (void)ws_size;
    const float* x[3]   = {(const float*)d_in[0],  (const float*)d_in[1],  (const float*)d_in[2]};
    const float* cw[3]  = {(const float*)d_in[3],  (const float*)d_in[7],  (const float*)d_in[11]};
    const float* cbv[3] = {(const float*)d_in[4],  (const float*)d_in[8],  (const float*)d_in[12]};
    const float* gg[3]  = {(const float*)d_in[5],  (const float*)d_in[9],  (const float*)d_in[13]};
    const float* gb[3]  = {(const float*)d_in[6],  (const float*)d_in[10], (const float*)d_in[14]};
    const float* pw1 = (const float*)d_in[15];
    const float* pb1 = (const float*)d_in[16];
    const float* pw2 = (const float*)d_in[17];
    const float* pb2 = (const float*)d_in[18];
    const float* lng = (const float*)d_in[19];
    const float* lnb = (const float*)d_in[20];
    const float* qw1 = (const float*)d_in[21];
    const float* qb1 = (const float*)d_in[22];
    const float* qw2 = (const float*)d_in[23];
    const float* qb2 = (const float*)d_in[24];

    float* out  = (float*)d_out;
    float* enc  = out + ENC_OFF;
    float* proj = out + PROJ_OFF;
    float* pred = out + PRED_OFF;
    float* loss = out + LOSS_OFF;

    unsigned* wpack   = (unsigned*)d_ws;
    float* inv_norm   = (float*)((char*)d_ws + WPACK_BYTES);
    float* rowloss    = inv_norm + 768;

    // 0) pack conv weights into WMMA fragment order (bf16)
    pack_weights_kernel<<<(WPACK_DW + 255) / 256, 256, 0, stream>>>(cw[0], cw[1], cw[2], wpack);

    // 1) fused conv + groupnorm + relu + pool (WMMA bf16)
    ConvArgs ca;
    for (int m = 0; m < 3; ++m) { ca.x[m] = x[m]; ca.cbias[m] = cbv[m]; ca.gng[m] = gg[m]; ca.gnb[m] = gb[m]; }
    conv_gn_pool_kernel<<<768, 256, SMEM_BYTES, stream>>>(ca, wpack, enc);

    // 2) projector
    projector_kernel<<<768, 256, 0, stream>>>(enc, pw1, pb1, pw2, pb2, lng, lnb, proj);

    // 3) predictor
    predictor_kernel<<<768, 128, 0, stream>>>(proj, qw1, qb1, qw2, qb2, pred);

    // 4) loss
    norms_kernel<<<3, 256, 0, stream>>>(proj, inv_norm);
    loss_row_kernel<<<768, 256, 0, stream>>>(proj, inv_norm, rowloss);
    loss_reduce_kernel<<<1, 32, 0, stream>>>(rowloss, loss);
}
